// CoModConv_88974542504488
// MI455X (gfx1250) — compile-verified, matrix operand order checked by reading, not run on hardware
//
#include <hip/hip_runtime.h>

// ---------------------------------------------------------------------------
// CoModConv for MI455X (gfx1250): style-MLP -> per-channel modulation ->
// 3x3 conv as implicit GEMM on v_wmma_f32_16x16x32_bf16 (f32 accumulate).
//
// x_mod is stored bf16 NHWC with a 1-pixel zero halo (16 x 66 x 66 x 256,
// ~35.7MB) so the conv inner loop is completely branch-free (EXEC all-1s).
// It and the repacked bf16 weights (1.2MB) are L2-resident (192MB), so the
// GEMM streams operands with global_load_b128 straight from L2.
// Wave tile 32co x 32px (2x2 WMMA tiles); the 72-step K loop (9 taps x 8
// chunks) is fully unrolled and software-pipelined one step ahead: loads for
// step ks+1 issue before the 4 WMMAs of step ks, so the matrix pipe never
// waits a full L2 round-trip.
// ---------------------------------------------------------------------------

typedef __bf16 bf16_t;
typedef __attribute__((ext_vector_type(8)))  __bf16 v8bf;
typedef __attribute__((ext_vector_type(16))) __bf16 v16bf;
typedef __attribute__((ext_vector_type(8)))  float  v8f;

#define BATCH 16
#define DCAT  512
#define CIN   256
#define COUT  256
#define HH    64
#define WW    64
#define HP    66          // padded height
#define WP    66          // padded width

static __device__ __forceinline__ float lrelu(float v) {
    return v >= 0.0f ? v : 0.01f * v;
}

static __device__ __forceinline__ v16bf cat16(v8bf lo, v8bf hi) {
    return __builtin_shufflevector(lo, hi,
        0,1,2,3,4,5,6,7,8,9,10,11,12,13,14,15);
}

static __device__ __forceinline__ v8f wmma_bf16(v16bf a, v16bf b, v8f c) {
    return __builtin_amdgcn_wmma_f32_16x16x32_bf16(
        false, a, false, b, (short)0, c, false, false);
}

// --------------------------- 1) style MLP ----------------------------------
__global__ __launch_bounds__(256)
void style_mlp_kernel(const float* __restrict__ y,
                      const float* __restrict__ w0, const float* __restrict__ b0,
                      const float* __restrict__ w1, const float* __restrict__ b1,
                      const float* __restrict__ w2, const float* __restrict__ b2,
                      float* __restrict__ s_out) {
    __shared__ float sb0[BATCH * CIN];
    __shared__ float sb1[BATCH * CIN];
    const int c = threadIdx.x;  // one output channel per thread

    for (int b = 0; b < BATCH; ++b) {
        float acc = b0[c];
        const float* wr = w0 + c * DCAT;
        const float* yr = y + b * DCAT;
        #pragma unroll 4
        for (int k = 0; k < DCAT; ++k) acc += yr[k] * wr[k];
        sb0[b * CIN + c] = lrelu(acc);
    }
    __syncthreads();
    for (int b = 0; b < BATCH; ++b) {
        float acc = b1[c];
        const float* wr = w1 + c * CIN;
        const float* sr = &sb0[b * CIN];
        #pragma unroll 4
        for (int k = 0; k < CIN; ++k) acc += sr[k] * wr[k];
        sb1[b * CIN + c] = lrelu(acc);
    }
    __syncthreads();
    for (int b = 0; b < BATCH; ++b) {
        float acc = b2[c];
        const float* wr = w2 + c * CIN;
        const float* sr = &sb1[b * CIN];
        #pragma unroll 4
        for (int k = 0; k < CIN; ++k) acc += sr[k] * wr[k];
        s_out[b * CIN + c] = lrelu(acc);
    }
}

// --------------------------- 2) weight repack -------------------------------
// conv_w (co, ci, kh, kw) f32  ->  wb[(kh*3+kw)][co][ci] bf16 (K=ci contiguous)
__global__ __launch_bounds__(256)
void repack_w_kernel(const float* __restrict__ cw, bf16_t* __restrict__ wb) {
    const int idx = blockIdx.x * 256 + threadIdx.x;   // 9*256*256
    if (idx >= 9 * COUT * CIN) return;
    const int ci = idx & (CIN - 1);
    const int co = (idx >> 8) & (COUT - 1);
    const int kk = idx >> 16;          // 0..8
    const int kh = kk / 3, kw = kk % 3;
    wb[idx] = (bf16_t)cw[(((co * CIN) + ci) * 3 + kh) * 3 + kw];
}

// --------------------------- 3a) zero-fill padded buffer --------------------
__global__ __launch_bounds__(256)
void zero_fill_kernel(v8bf* __restrict__ p) {
    const v8bf z = {};
    p[blockIdx.x * 256 + threadIdx.x] = z;
}

// --------------------------- 3b) modulate + NCHW->NHWC(bf16, halo) ----------
// xb[b][h+1][w+1][c] = bf16(x[b][c][h][w] * s[b][c]); LDS transpose tile
__global__ __launch_bounds__(256)
void modulate_kernel(const float* __restrict__ x, const float* __restrict__ s,
                     bf16_t* __restrict__ xb) {
    __shared__ float tile[WW][33];     // [w][c], padded
    const int cb = blockIdx.x & 7;          // channel block of 32
    const int h  = (blockIdx.x >> 3) & 63;
    const int b  = blockIdx.x >> 9;
    const int t  = threadIdx.x;

    #pragma unroll
    for (int i = 0; i < 8; ++i) {
        const int e  = t + i * 256;
        const int c  = e >> 6;              // 0..31
        const int w  = e & 63;
        const int ch = cb * 32 + c;
        tile[w][c] = x[(((b * CIN) + ch) * HH + h) * WW + w] * s[b * CIN + ch];
    }
    __syncthreads();
    const int w  = t >> 2;
    const int cj = (t & 3) * 8;
    v8bf pkt;
    #pragma unroll
    for (int i = 0; i < 8; ++i) pkt[i] = (bf16_t)tile[w][cj + i];
    *(v8bf*)(xb + (size_t)((b * HP + h + 1) * WP + w + 1) * CIN
                + cb * 32 + cj) = pkt;
}

// --------------------------- 4) implicit-GEMM conv --------------------------
// Block: 256 thr = 8 waves; block tile M=128(co) x N=64(px) = one image row.
// Wave = (mwave 0..3) x (nwave 0..1): 32co x 32px, accumulators 2x2.
// K loop: 72 steps = 9 taps x 8 channel-chunks, fully unrolled, software-
// pipelined one step ahead (double-buffered operand registers).
__global__ __launch_bounds__(256)
void comod_conv_kernel(const bf16_t* __restrict__ xb,   // NHWC bf16, haloed
                       const bf16_t* __restrict__ wb,   // [kk][co][ci] bf16
                       float* __restrict__ out) {       // NCHW f32
    const int nb = blockIdx.x;            // 0..1023 -> (b, h)
    const int b  = nb >> 6;
    const int h  = nb & 63;
    const int mblk = blockIdx.y;          // 0..1 -> co base 128*mblk

    const int tid   = threadIdx.x;
    const int lane  = tid & 31;
    const int wid   = tid >> 5;
    const int mwave = wid & 3;            // 0..3
    const int nwave = wid >> 2;           // 0..1
    const int w0p   = nwave * 32;         // pixel base in row
    const int cobase = mblk * 128 + mwave * 32;

    const int lrow  = lane & 15;          // A row / B column within tile
    const int khalf = lane >> 4;          // K-half select (16-bit layout)
    const int koff  = khalf * 8;

    v8f acc00 = {}, acc01 = {}, acc10 = {}, acc11 = {};

    const int px0 = w0p + lrow;           // n-tile0 pixel col (unpadded w)
    // padded coords: input row = h + kh (0..65), input col = px + kw (0..65)
    const bf16_t* brow = xb + (size_t)(b * HP + h) * WP * CIN + koff;
    const bf16_t* arow = wb + (size_t)(cobase + lrow) * CIN + koff;

    // ks (0..71) is a compile-time constant under full unroll, so these fold
    // into immediate offsets off two base address registers.
    auto a0p = [&](int ks) {
        return arow + ((ks >> 3) * COUT * CIN) + ((ks & 7) * 32);
    };
    auto b0p = [&](int ks) {
        const int tap = ks >> 3;
        return brow + (size_t)((tap / 3) * WP + px0 + (tap % 3)) * CIN
                    + ((ks & 7) * 32);
    };

    // prologue: load step 0 operands
    const bf16_t* pa0 = a0p(0);
    const bf16_t* pb0 = b0p(0);
    v8bf ca0l = *(const v8bf*)(pa0);
    v8bf ca0h = *(const v8bf*)(pa0 + 16);
    v8bf ca1l = *(const v8bf*)(pa0 + 16 * CIN);
    v8bf ca1h = *(const v8bf*)(pa0 + 16 * CIN + 16);
    v8bf cb0l = *(const v8bf*)(pb0);
    v8bf cb0h = *(const v8bf*)(pb0 + 16);
    v8bf cb1l = *(const v8bf*)(pb0 + 16 * CIN);
    v8bf cb1h = *(const v8bf*)(pb0 + 16 * CIN + 16);

    #pragma unroll
    for (int ks = 0; ks < 72; ++ks) {
        // issue next step's loads before this step's WMMAs
        v8bf na0l = {}, na0h = {}, na1l = {}, na1h = {};
        v8bf nb0l = {}, nb0h = {}, nb1l = {}, nb1h = {};
        if (ks + 1 < 72) {
            const bf16_t* qa = a0p(ks + 1);
            const bf16_t* qb = b0p(ks + 1);
            na0l = *(const v8bf*)(qa);
            na0h = *(const v8bf*)(qa + 16);
            na1l = *(const v8bf*)(qa + 16 * CIN);
            na1h = *(const v8bf*)(qa + 16 * CIN + 16);
            nb0l = *(const v8bf*)(qb);
            nb0h = *(const v8bf*)(qb + 16);
            nb1l = *(const v8bf*)(qb + 16 * CIN);
            nb1h = *(const v8bf*)(qb + 16 * CIN + 16);
        }

        const v16bf A0 = cat16(ca0l, ca0h);
        const v16bf A1 = cat16(ca1l, ca1h);
        const v16bf B0 = cat16(cb0l, cb0h);
        const v16bf B1 = cat16(cb1l, cb1h);
        acc00 = wmma_bf16(A0, B0, acc00);
        acc01 = wmma_bf16(A0, B1, acc01);
        acc10 = wmma_bf16(A1, B0, acc10);
        acc11 = wmma_bf16(A1, B1, acc11);

        // rotate double buffers (renamed away by SSA under full unroll)
        ca0l = na0l; ca0h = na0h; ca1l = na1l; ca1h = na1h;
        cb0l = nb0l; cb0h = nb0h; cb1l = nb1l; cb1h = nb1h;
    }

    // Epilogue: C/D layout -> lane L, reg r: co = base + 8*(L/16) + r, px=L%16
    const int mrow = khalf * 8;
    #pragma unroll
    for (int r = 0; r < 8; ++r) {
        const int co = cobase + mrow + r;
        const size_t base0 = ((size_t)(b * COUT + co) * HH + h) * WW;
        out[base0 + px0]      = acc00[r];
        out[base0 + px0 + 16] = acc01[r];
        const size_t base1 = ((size_t)(b * COUT + co + 16) * HH + h) * WW;
        out[base1 + px0]      = acc10[r];
        out[base1 + px0 + 16] = acc11[r];
    }
}

// ---------------------------------------------------------------------------
extern "C" void kernel_launch(void* const* d_in, const int* in_sizes, int n_in,
                              void* d_out, int out_size, void* d_ws, size_t ws_size,
                              hipStream_t stream) {
    const float* x      = (const float*)d_in[0];
    const float* y      = (const float*)d_in[1];
    const float* w0     = (const float*)d_in[2];
    const float* b0     = (const float*)d_in[3];
    const float* w1     = (const float*)d_in[4];
    const float* b1     = (const float*)d_in[5];
    const float* w2     = (const float*)d_in[6];
    const float* b2     = (const float*)d_in[7];
    const float* conv_w = (const float*)d_in[8];
    float* out          = (float*)d_out;

    // ws layout: s (16KB f32) | xb (haloed NHWC bf16, 35.7MB) | wb (bf16 1.2MB)
    const size_t xb_bytes = (size_t)BATCH * HP * WP * CIN * 2;
    char* ws      = (char*)d_ws;
    float*  s     = (float*)ws;
    bf16_t* xbuf  = (bf16_t*)(ws + 16384);
    bf16_t* wbuf  = (bf16_t*)(ws + 16384 + xb_bytes);

    style_mlp_kernel<<<1, 256, 0, stream>>>(y, w0, b0, w1, b1, w2, b2, s);
    repack_w_kernel<<<(9 * COUT * CIN) / 256, 256, 0, stream>>>(conv_w, wbuf);
    // zero the whole haloed buffer (v8bf granules), then write the interior
    const int fill_blocks = (int)(xb_bytes / 16 / 256);   // 8712
    zero_fill_kernel<<<fill_blocks, 256, 0, stream>>>((v8bf*)xbuf);
    modulate_kernel<<<BATCH * HH * (CIN / 32), 256, 0, stream>>>(x, s, xbuf);
    comod_conv_kernel<<<dim3(BATCH * HH, COUT / 128), 256, 0, stream>>>(xbuf, wbuf, out);
}
